// PointNet2EncoderLocal_35485019799952
// MI455X (gfx1250) — matrix operand order repaired and checked by reading, not compiled, not run on hardware
//
#include <hip/hip_runtime.h>
#include <hip/hip_bf16.h>
#include <cstdint>
#include <cstddef>

// ---------------------------------------------------------------------------
// PointNet++ (MSG) encoder forward for MI455X / gfx1250 (wave32, WMMA).
// All MLP layers run as WMMA f16->f32 GEMMs with padded f16 layouts:
// branch-free inner loop = 4x global_load_b128 + 1x v_wmma per K=32 step.
// ---------------------------------------------------------------------------

typedef __attribute__((ext_vector_type(16))) _Float16 v16h;
typedef __attribute__((ext_vector_type(8)))  _Float16 v8h;
typedef __attribute__((ext_vector_type(8)))  float    v8f;

#define WPB 8   // waves per block (256 threads, wave32)

static inline int rup32(int x) { return (x + 31) & ~31; }
static inline int blks(long long n, int t) { return (int)((n + t - 1) / t); }

// ----------------------------- weight/param packing (zero-padded, f16)
__global__ void k_prep_w(const float* __restrict__ W, _Float16* __restrict__ Wp,
                         int cin, int cout, int cinp, int coutp) {
  int id = blockIdx.x * blockDim.x + threadIdx.x;
  if (id >= coutp * cinp) return;
  int o = id / cinp, k = id % cinp;
  float v = (o < cout && k < cin) ? W[(size_t)o * cin + k] : 0.f;
  Wp[id] = (_Float16)v;
}

__global__ void k_prep_v(const float* __restrict__ b, const float* __restrict__ g,
                         const float* __restrict__ bt,
                         float* __restrict__ bp, float* __restrict__ gp,
                         float* __restrict__ btp, int cout, int coutp) {
  int id = blockIdx.x * blockDim.x + threadIdx.x;
  if (id >= coutp) return;
  bool in = id < cout;
  bp[id]  = in ? b[id]  : 0.f;
  gp[id]  = in ? g[id]  : 0.f;   // gamma=0 on pad -> output 0 -> keeps K-pad zero
  btp[id] = in ? bt[id] : 0.f;
}

// ----------------------------- transpose (B,3,N) -> (B,N,3) f32
__global__ void k_transpose_in(const float* __restrict__ in, float* __restrict__ out,
                               int B, int N) {
  int id = blockIdx.x * blockDim.x + threadIdx.x;
  if (id >= B * N) return;
  int b = id / N, n = id % N;
  const float* s = in + (size_t)b * 3 * N + n;
  float* d = out + (size_t)id * 3;
  d[0] = s[0]; d[1] = s[(size_t)N]; d[2] = s[(size_t)2 * N];
}

// ----------------------------- farthest point sampling (1 block / batch)
__global__ void k_fps(const float* __restrict__ xyz, int* __restrict__ outIdx,
                      int N, int npoint) {
  __shared__ float dist[2048];
  __shared__ float rv[256];
  __shared__ int   ri[256];
  __shared__ int   sFar;
  const int b   = blockIdx.x;
  const int tid = threadIdx.x;
  const float* P = xyz + (size_t)b * N * 3;
  for (int n = tid; n < N; n += blockDim.x) dist[n] = 1e10f;
  if (tid == 0) sFar = 0;
  __syncthreads();
  for (int it = 0; it < npoint; ++it) {
    int far = sFar;
    if (tid == 0) outIdx[b * npoint + it] = far;
    float cx = P[far * 3 + 0], cy = P[far * 3 + 1], cz = P[far * 3 + 2];
    float bv = -1.f; int bi = 0;
    for (int n = tid; n < N; n += blockDim.x) {
      float dx = P[n * 3 + 0] - cx, dy = P[n * 3 + 1] - cy, dz = P[n * 3 + 2] - cz;
      float d = dx * dx + dy * dy + dz * dz;
      float dm = dist[n]; dm = d < dm ? d : dm; dist[n] = dm;
      if (dm > bv) { bv = dm; bi = n; }
    }
    rv[tid] = bv; ri[tid] = bi;
    __syncthreads();
    for (int s = 128; s > 0; s >>= 1) {
      if (tid < s) {
        float v2 = rv[tid + s]; int i2 = ri[tid + s];
        if (v2 > rv[tid] || (v2 == rv[tid] && i2 < ri[tid])) { rv[tid] = v2; ri[tid] = i2; }
      }
      __syncthreads();
    }
    if (tid == 0) sFar = ri[0];
    __syncthreads();
  }
}

// ----------------------------- gather 3-vectors by index (f32)
__global__ void k_gather3(const float* __restrict__ src, const int* __restrict__ idx,
                          float* __restrict__ dst, int B, int N, int S) {
  int id = blockIdx.x * blockDim.x + threadIdx.x;
  if (id >= B * S) return;
  int b = id / S;
  int j = idx[id];
  const float* s = src + ((size_t)b * N + j) * 3;
  float* d = dst + (size_t)id * 3;
  d[0] = s[0]; d[1] = s[1]; d[2] = s[2];
}

// ----------------------------- ball query: one wave per query point (wave32)
__global__ void k_ballquery(const float* __restrict__ xyz, const float* __restrict__ nxyz,
                            int* __restrict__ out, int B, int N, int S, int K, float r2) {
  __shared__ int buf[WPB][128];
  const int lane = threadIdx.x & 31;
  const int w    = threadIdx.x >> 5;
  const int q    = blockIdx.x * WPB + w;
  const bool active = q < B * S;
  int cnt = 0;
  if (active) {
    int b = q / S;
    const float* C = nxyz + (size_t)q * 3;
    const float cx = C[0], cy = C[1], cz = C[2];
    const float* P = xyz + (size_t)b * N * 3;
    for (int base = 0; base < N && cnt < K; base += 32) {
      int n = base + lane;
      bool in = false;
      if (n < N) {
        float dx = P[n * 3 + 0] - cx, dy = P[n * 3 + 1] - cy, dz = P[n * 3 + 2] - cz;
        in = (dx * dx + dy * dy + dz * dz) <= r2;
      }
      unsigned mm = (unsigned)__ballot(in);
      int rank = __popc(mm & ((1u << lane) - 1u));
      if (in && cnt + rank < K) buf[w][cnt + rank] = n;
      cnt += __popc(mm);
    }
  }
  __syncthreads();
  if (active) {
    int c = cnt < K ? cnt : K;
    int fill = (c == 0) ? (N - 1) : buf[w][0];
    for (int j = lane; j < K; j += 32) {
      out[(size_t)q * K + j] = (j < c) ? buf[w][j] : fill;
    }
  }
}

// ----------------------------- grouped features -> f16, padded ld, zero pad
__global__ void k_group(const float* __restrict__ xyz, const _Float16* __restrict__ pts,
                        int pts_is_xyz, const float* __restrict__ nxyz,
                        const int* __restrict__ idx, _Float16* __restrict__ G,
                        int S, int K, int Cp, int Cinp) {
  size_t id = (size_t)blockIdx.x * blockDim.x + threadIdx.x;
  size_t total = (size_t)S * K * Cinp;
  if (id >= total) return;
  int c = (int)(id % Cinp);
  size_t r = id / Cinp;
  int s = (int)(r / K);
  int j = idx[r];
  float v;
  if (c < Cp)          v = pts_is_xyz ? xyz[(size_t)j * 3 + c]
                                      : (float)pts[(size_t)j * Cp + c];
  else if (c < Cp + 3) { int d = c - Cp; v = xyz[(size_t)j * 3 + d] - nxyz[(size_t)s * 3 + d]; }
  else                 v = 0.f;
  G[id] = (_Float16)v;
}

// ----------------------------- WMMA GEMM + affine + ReLU (all padded, branch-free)
// Y(M x Np) = relu(gamma * (X(M x Kp) * W(Np x Kp)^T + b) + beta), f16 in/out, f32 acc
__global__ __launch_bounds__(256)
void k_gemm(const _Float16* __restrict__ X, const _Float16* __restrict__ W,
            const float* __restrict__ bias, const float* __restrict__ gam,
            const float* __restrict__ bet, _Float16* __restrict__ Y,
            int M, int Kp, int Np) {
  const int lane   = threadIdx.x & 31;
  const int wv     = threadIdx.x >> 5;
  const int tilesN = Np >> 4;
  const int tile   = blockIdx.x * WPB + wv;
  if (tile >= (M >> 4) * tilesN) return;     // wave-uniform: EXEC stays all-ones
  const int m0   = (tile / tilesN) << 4;
  const int n0   = (tile % tilesN) << 4;
  const int half = lane >> 4;
  const int l16  = lane & 15;

  // Fragment layouts: per lane-half, A and B both cover contiguous [16h, 16h+16).
  const _Float16* __restrict__ Xr = X + (size_t)(m0 + l16) * Kp + 16 * half;
  const _Float16* __restrict__ Wr = W + (size_t)(n0 + l16) * Kp + 16 * half;

  v8f acc = {};
#pragma unroll 4
  for (int k0 = 0; k0 < Kp; k0 += 32) {
    v16h a  = *(const v16h*)(Xr + k0);
    v16h b  = *(const v16h*)(Wr + k0);
    acc = __builtin_amdgcn_wmma_f32_16x16x32_f16(false, a, false, b, (short)0, acc,
                                                 false, false);
  }

  const int n = n0 + l16;
  const float bb = bias[n], g = gam[n], bt = bet[n];
  _Float16* Yc = Y + n;
#pragma unroll
  for (int r = 0; r < 8; ++r) {
    const int m = m0 + r + 8 * half;   // C/D layout: VGPR r, lane-half selects +8
    float v = g * (acc[r] + bb) + bt;
    v = v > 0.f ? v : 0.f;
    Yc[(size_t)m * Np] = (_Float16)v;
  }
}

// ----------------------------- max over K rows per group (f16)
__global__ void k_maxpool(const _Float16* __restrict__ H, _Float16* __restrict__ O,
                          int S, int K, int C, int ldh, int ldo, int co0) {
  int id = blockIdx.x * blockDim.x + threadIdx.x;
  if (id >= S * C) return;
  int s = id / C, c = id % C;
  const _Float16* h = H + (size_t)s * K * ldh + c;
  float m = -3.4e38f;
  for (int k = 0; k < K; ++k) { float v = (float)h[(size_t)k * ldh]; m = v > m ? v : m; }
  O[(size_t)s * ldo + co0 + c] = (_Float16)m;
}

// ----------------------------- group_all concat [xyz | pts] -> f16 padded
__global__ void k_groupall(const float* __restrict__ xyz, const _Float16* __restrict__ pts,
                           _Float16* __restrict__ G, int B, int Np, int Cp, int Cinp) {
  size_t id = (size_t)blockIdx.x * blockDim.x + threadIdx.x;
  size_t total = (size_t)B * Np * Cinp;
  if (id >= total) return;
  int c = (int)(id % Cinp);
  size_t row = id / Cinp;
  float v;
  if (c < 3)           v = xyz[row * 3 + c];
  else if (c < 3 + Cp) v = (float)pts[(size_t)row * Cp + (c - 3)];
  else                 v = 0.f;
  G[id] = (_Float16)v;
}

// ----------------------------- fp3 concat: [l2_points | broadcast l3] -> f16
__global__ void k_concat_fp3(const _Float16* __restrict__ l2p, const _Float16* __restrict__ l3,
                             _Float16* __restrict__ G, int B, int Np, int C1, int C2) {
  int Cin = C1 + C2;
  size_t total = (size_t)B * Np * Cin;
  size_t id = (size_t)blockIdx.x * blockDim.x + threadIdx.x;
  if (id >= total) return;
  int c = (int)(id % Cin);
  size_t row = id / Cin;
  int b = (int)(row / Np);
  G[id] = (c < C1) ? l2p[(size_t)row * C1 + c] : l3[(size_t)b * C2 + (c - C1)];
}

// ----------------------------- 3-NN + inverse-distance weights (f32 geometry)
__global__ void k_three_nn(const float* __restrict__ q, const float* __restrict__ ref,
                           int* __restrict__ oi, float* __restrict__ ow,
                           int B, int Nq, int Nr) {
  int id = blockIdx.x * blockDim.x + threadIdx.x;
  if (id >= B * Nq) return;
  int b = id / Nq;
  const float qx = q[(size_t)id * 3 + 0], qy = q[(size_t)id * 3 + 1], qz = q[(size_t)id * 3 + 2];
  const float* R = ref + (size_t)b * Nr * 3;
  float d0 = 3.0e38f, d1 = 3.0e38f, d2 = 3.0e38f;
  int   i0 = 0, i1 = 0, i2 = 0;
  for (int n = 0; n < Nr; ++n) {
    float dx = R[n * 3 + 0] - qx, dy = R[n * 3 + 1] - qy, dz = R[n * 3 + 2] - qz;
    float d = dx * dx + dy * dy + dz * dz;
    if (d < d0)      { d2 = d1; i2 = i1; d1 = d0; i1 = i0; d0 = d; i0 = n; }
    else if (d < d1) { d2 = d1; i2 = i1; d1 = d; i1 = n; }
    else if (d < d2) { d2 = d; i2 = n; }
  }
  float w0 = 1.f / (d0 + 1e-8f), w1 = 1.f / (d1 + 1e-8f), w2 = 1.f / (d2 + 1e-8f);
  float s = w0 + w1 + w2;
  oi[(size_t)id * 3 + 0] = i0; oi[(size_t)id * 3 + 1] = i1; oi[(size_t)id * 3 + 2] = i2;
  ow[(size_t)id * 3 + 0] = w0 / s; ow[(size_t)id * 3 + 1] = w1 / s; ow[(size_t)id * 3 + 2] = w2 / s;
}

// ----------------------------- [p1 | 3-NN interp of p2] -> f16 padded
__global__ void k_interp_concat(const _Float16* __restrict__ p1,
                                const float* __restrict__ xyzdup, int C1,
                                const _Float16* __restrict__ p2, int C2, int Np2,
                                const int* __restrict__ nni, const float* __restrict__ nnw,
                                _Float16* __restrict__ G, int B, int Nq, int Cinp) {
  size_t total = (size_t)B * Nq * Cinp;
  size_t id = (size_t)blockIdx.x * blockDim.x + threadIdx.x;
  if (id >= total) return;
  int c = (int)(id % Cinp);
  size_t row = id / Cinp;
  int b = (int)(row / Nq);
  float v;
  if (c < C1) {
    v = xyzdup ? xyzdup[row * 3 + (c % 3)] : (float)p1[row * (size_t)C1 + c];
  } else if (c < C1 + C2) {
    int cc = c - C1;
    const int*   ii = nni + row * 3;
    const float* ww = nnw + row * 3;
    v = 0.f;
#pragma unroll
    for (int t = 0; t < 3; ++t)
      v += ww[t] * (float)p2[((size_t)b * Np2 + ii[t]) * C2 + cc];
  } else {
    v = 0.f;
  }
  G[id] = (_Float16)v;
}

// ----------------------------- final transpose (B,N,C) f16 -> (B,C,N) f32
__global__ void k_transpose_out(const _Float16* __restrict__ H, float* __restrict__ out,
                                int B, int N, int C) {
  size_t total = (size_t)B * N * C;
  size_t id = (size_t)blockIdx.x * blockDim.x + threadIdx.x;
  if (id >= total) return;
  int b = (int)(id / ((size_t)N * C));
  int rest = (int)(id % ((size_t)N * C));
  int n = rest / C, c = rest % C;
  out[(size_t)b * C * N + (size_t)c * N + n] = (float)H[id];
}

// ===========================================================================
// Host orchestration
// ===========================================================================

struct Lay {
  const float *W, *b, *beta, *gamma;   // original f32 leaves
  int cin, cout, cinp, coutp;
  _Float16* Wp;                        // packed padded f16
  float *bp, *gp, *btp;                // padded bias/gamma/beta
};

static void gemm(const _Float16* X, const Lay& L, _Float16* Y, int M, hipStream_t s) {
  int tiles = (M >> 4) * (L.coutp >> 4);
  k_gemm<<<blks(tiles, WPB), 256, 0, s>>>(X, L.Wp, L.bp, L.gp, L.btp, Y,
                                          M, L.cinp, L.coutp);
}

extern "C" void kernel_launch(void* const* d_in, const int* in_sizes, int n_in,
                              void* d_out, int out_size, void* d_ws, size_t ws_size,
                              hipStream_t stream) {
  const int B = 8, N = 2048;
  (void)ws_size; (void)out_size;

  // Locate xyz (49152 elements) vs. the 96 parameter leaves.
  const float* xyz_in;
  void* const* P;
  if (in_sizes[0] == B * 3 * N) { xyz_in = (const float*)d_in[0]; P = d_in + 1; }
  else                          { xyz_in = (const float*)d_in[n_in - 1]; P = d_in; }

  // Param leaf order: jax tree_flatten (sorted dict keys); per layer W,b,beta,gamma.
  // {leaf base, cin, cout} in flatten order: fp1, fp2, fp3, sa1(x3), sa2(x2), sa3.
  static const int defs[24][3] = {
    {  0, 134, 128 }, {  4, 128, 128 },                       // fp1
    {  8, 576, 256 }, { 12, 256, 128 },                       // fp2
    { 16, 1536, 256 }, { 20, 256, 256 },                      // fp3
    { 24, 6, 32 },  { 28, 32, 32 },  { 32, 32, 64 },          // sa1 b0
    { 36, 6, 64 },  { 40, 64, 64 },  { 44, 64, 128 },         // sa1 b1
    { 48, 6, 64 },  { 52, 64, 96 },  { 56, 96, 128 },         // sa1 b2
    { 60, 323, 128 }, { 64, 128, 128 }, { 68, 128, 256 },     // sa2 b0
    { 72, 323, 128 }, { 76, 128, 196 }, { 80, 196, 256 },     // sa2 b1
    { 84, 515, 256 }, { 88, 256, 512 }, { 92, 512, 1024 }     // sa3
  };

  // ---- workspace carve
  char* wsb = (char*)d_ws;
  size_t off = 0;
  auto alloc = [&](size_t nbytes) -> void* {
    void* p = wsb + off;
    off = (off + nbytes + 255) & ~((size_t)255);
    return p;
  };

  Lay L[24];
  for (int i = 0; i < 24; ++i) {
    int base = defs[i][0];
    L[i].W     = (const float*)P[base + 0];
    L[i].b     = (const float*)P[base + 1];
    L[i].beta  = (const float*)P[base + 2];
    L[i].gamma = (const float*)P[base + 3];
    L[i].cin = defs[i][1]; L[i].cout = defs[i][2];
    L[i].cinp = rup32(L[i].cin); L[i].coutp = rup32(L[i].cout);
    L[i].Wp  = (_Float16*)alloc((size_t)L[i].coutp * L[i].cinp * 2);
    L[i].bp  = (float*)alloc((size_t)L[i].coutp * 4);
    L[i].gp  = (float*)alloc((size_t)L[i].coutp * 4);
    L[i].btp = (float*)alloc((size_t)L[i].coutp * 4);
  }

  float*     l0_xyz = (float*)alloc((size_t)B * N * 3 * 4);
  int*       fps1   = (int*)  alloc((size_t)B * 512 * 4);
  int*       fps2   = (int*)  alloc((size_t)B * 128 * 4);
  float*     l1_xyz = (float*)alloc((size_t)B * 512 * 3 * 4);
  _Float16*  l1_pts = (_Float16*)alloc((size_t)B * 512 * 320 * 2);
  float*     l2_xyz = (float*)alloc((size_t)B * 128 * 3 * 4);
  _Float16*  l2_pts = (_Float16*)alloc((size_t)B * 128 * 512 * 2);
  _Float16*  l3_pts = (_Float16*)alloc((size_t)B * 1024 * 2);
  _Float16*  l2_fp  = (_Float16*)alloc((size_t)B * 128 * 256 * 2);
  _Float16*  l1_fp  = (_Float16*)alloc((size_t)B * 512 * 128 * 2);
  int*       idxb   = (int*)  alloc((size_t)B * 512 * 128 * 4);
  int*       nni    = (int*)  alloc((size_t)B * 2048 * 3 * 4);
  float*     nnw    = (float*)alloc((size_t)B * 2048 * 3 * 4);
  _Float16*  pingA  = (_Float16*)alloc((size_t)6291456 * 2);  // 65536x96 / 16384x352
  _Float16*  pingB  = (_Float16*)alloc((size_t)8388608 * 2);  // 65536x128

  // ---- pack weights (f16, zero padded) and bias/gamma/beta
  for (int i = 0; i < 24; ++i) {
    k_prep_w<<<blks((long long)L[i].coutp * L[i].cinp, 256), 256, 0, stream>>>(
        L[i].W, L[i].Wp, L[i].cin, L[i].cout, L[i].cinp, L[i].coutp);
    k_prep_v<<<blks(L[i].coutp, 256), 256, 0, stream>>>(
        L[i].b, L[i].gamma, L[i].beta, L[i].bp, L[i].gp, L[i].btp,
        L[i].cout, L[i].coutp);
  }

  // ---- l0
  k_transpose_in<<<blks(B * N, 256), 256, 0, stream>>>(xyz_in, l0_xyz, B, N);

  // ---- SA1 (npoint=512, radii .1/.2/.4, K 32/64/128, out 64+128+128=320)
  k_fps<<<B, 256, 0, stream>>>(l0_xyz, fps1, N, 512);
  k_gather3<<<blks(B * 512, 256), 256, 0, stream>>>(l0_xyz, fps1, l1_xyz, B, N, 512);
  {
    const int   Ks[3]   = { 32, 64, 128 };
    const float r2s[3]  = { 0.01f, 0.04f, 0.16f };
    const int   coff[3] = { 0, 64, 192 };
    for (int br = 0; br < 3; ++br) {
      int K = Ks[br];
      const Lay* ml = &L[6 + 3 * br];
      k_ballquery<<<blks(B * 512, WPB), 256, 0, stream>>>(l0_xyz, l1_xyz, idxb,
                                                          B, N, 512, K, r2s[br]);
      for (int b = 0; b < B; ++b) {
        const float* xb = l0_xyz + (size_t)b * N * 3;
        const float* nx = l1_xyz + (size_t)b * 512 * 3;
        const int*   ib = idxb + (size_t)b * 512 * K;
        int rows = 512 * K;
        k_group<<<blks((long long)rows * ml[0].cinp, 256), 256, 0, stream>>>(
            xb, (const _Float16*)nullptr, 1, nx, ib, pingA, 512, K, 3, ml[0].cinp);
        gemm(pingA, ml[0], pingB, rows, stream);
        gemm(pingB, ml[1], pingA, rows, stream);
        gemm(pingA, ml[2], pingB, rows, stream);
        k_maxpool<<<blks(512 * ml[2].cout, 256), 256, 0, stream>>>(
            pingB, l1_pts + (size_t)b * 512 * 320, 512, K, ml[2].cout, ml[2].coutp,
            320, coff[br]);
      }
    }
  }

  // ---- SA2 (npoint=128, radii .4/.8, K 64/128, out 256+256=512)
  k_fps<<<B, 256, 0, stream>>>(l1_xyz, fps2, 512, 128);
  k_gather3<<<blks(B * 128, 256), 256, 0, stream>>>(l1_xyz, fps2, l2_xyz, B, 512, 128);
  {
    const int   Ks[2]   = { 64, 128 };
    const float r2s[2]  = { 0.16f, 0.64f };
    const int   coff[2] = { 0, 256 };
    for (int br = 0; br < 2; ++br) {
      int K = Ks[br];
      const Lay* ml = &L[15 + 3 * br];
      k_ballquery<<<blks(B * 128, WPB), 256, 0, stream>>>(l1_xyz, l2_xyz, idxb,
                                                          B, 512, 128, K, r2s[br]);
      for (int b = 0; b < B; ++b) {
        const float*    xb = l1_xyz + (size_t)b * 512 * 3;
        const _Float16* pb = l1_pts + (size_t)b * 512 * 320;
        const float*    nx = l2_xyz + (size_t)b * 128 * 3;
        const int*      ib = idxb + (size_t)b * 128 * K;
        int rows = 128 * K;
        k_group<<<blks((long long)rows * ml[0].cinp, 256), 256, 0, stream>>>(
            xb, pb, 0, nx, ib, pingA, 128, K, 320, ml[0].cinp);
        gemm(pingA, ml[0], pingB, rows, stream);
        gemm(pingB, ml[1], pingA, rows, stream);
        gemm(pingA, ml[2], pingB, rows, stream);
        k_maxpool<<<blks(128 * ml[2].cout, 256), 256, 0, stream>>>(
            pingB, l2_pts + (size_t)b * 128 * 512, 128, K, ml[2].cout, ml[2].coutp,
            512, coff[br]);
      }
    }
  }

  // ---- SA3 group_all: concat([l2_xyz, l2_points]) -> 515 -> 256/512/1024 -> max
  k_groupall<<<blks((long long)B * 128 * L[21].cinp, 256), 256, 0, stream>>>(
      l2_xyz, l2_pts, pingA, B, 128, 512, L[21].cinp);
  gemm(pingA, L[21], pingB, B * 128, stream);
  gemm(pingB, L[22], pingA, B * 128, stream);
  gemm(pingA, L[23], pingB, B * 128, stream);
  k_maxpool<<<blks(B * 1024, 256), 256, 0, stream>>>(pingB, l3_pts, B, 128, 1024, 1024,
                                                     1024, 0);

  // ---- FP3: broadcast l3, concat with l2_points (512+1024=1536) -> 256 -> 256
  k_concat_fp3<<<blks((long long)B * 128 * 1536, 256), 256, 0, stream>>>(
      l2_pts, l3_pts, pingA, B, 128, 512, 1024);
  gemm(pingA, L[4], pingB, B * 128, stream);
  gemm(pingB, L[5], l2_fp, B * 128, stream);

  // ---- FP2: 3-NN l1<-l2, concat(320+256=576) -> 256 -> 128
  k_three_nn<<<blks(B * 512, 256), 256, 0, stream>>>(l1_xyz, l2_xyz, nni, nnw, B, 512, 128);
  k_interp_concat<<<blks((long long)B * 512 * 576, 256), 256, 0, stream>>>(
      l1_pts, (const float*)nullptr, 320, l2_fp, 256, 128, nni, nnw, pingA, B, 512, 576);
  gemm(pingA, L[2], pingB, B * 512, stream);
  gemm(pingB, L[3], l1_fp, B * 512, stream);

  // ---- FP1: 3-NN l0<-l1, concat([xyz,xyz]=6 + 128 = 134, pad 160) -> 128 -> 128
  k_three_nn<<<blks(B * 2048, 256), 256, 0, stream>>>(l0_xyz, l1_xyz, nni, nnw, B, 2048, 512);
  k_interp_concat<<<blks((long long)B * 2048 * L[0].cinp, 256), 256, 0, stream>>>(
      (const _Float16*)nullptr, l0_xyz, 6, l1_fp, 128, 512, nni, nnw, pingA, B, 2048,
      L[0].cinp);
  gemm(pingA, L[0], pingB, B * 2048, stream);
  gemm(pingB, L[1], pingA, B * 2048, stream);

  // ---- (B,N,128) f16 -> (B,128,N) f32
  k_transpose_out<<<blks((long long)B * 2048 * 128, 256), 256, 0, stream>>>(
      pingA, (float*)d_out, B, 2048, 128);
}
